// GATStochasticMuZeroModel_72971494359379
// MI455X (gfx1250) — compile-verified, matrix operand order checked by reading, not run on hardware
//
#include <hip/hip_runtime.h>

// ---------------------------------------------------------------------------
// GAT on 4x4 grid, B=4096. One wave32 per batch element, 8 waves per block.
// GEMMs via v_wmma_f32_16x16x32_bf16. Weights pre-swizzled (prep kernel) to
// B-fragment layout in d_ws, then cooperatively staged per block into LDS
// with global_load_async_to_lds_b128 (overlapped with attention compute),
// so all 8 waves share one L2 read of each layer's weights.
// ---------------------------------------------------------------------------

typedef __attribute__((ext_vector_type(16))) __bf16 v16bf;
typedef __attribute__((ext_vector_type(2)))  __bf16 v2bf;
typedef __attribute__((ext_vector_type(8)))  float  v8f;
typedef __attribute__((ext_vector_type(2)))  float  v2f;
typedef __attribute__((ext_vector_type(4)))  int    v4i;

#define WAVES_PER_BLOCK 8
#define THREADS 256
#define CH 256                       // max feature width (row stride in LDS)
#define LDS_PER_WAVE_BYTES 20480
#define WSTAGE_BYTES 131072          // 8*16*512 bf16 fragments (max layer)
#define SMEM_TOTAL (WSTAGE_BYTES + WAVES_PER_BLOCK * LDS_PER_WAVE_BYTES)

__device__ __forceinline__ unsigned short f2bf(float f) {
  unsigned u = __float_as_uint(f);
  u += 0x7FFFu + ((u >> 16) & 1u);        // round-to-nearest-even
  return (unsigned short)(u >> 16);
}
__device__ __forceinline__ float bf2f(unsigned short h) {
  return __uint_as_float(((unsigned)h) << 16);
}
// packed f32x2 -> bf16x2 (backend selects v_cvt_pk_bf16_f32 on gfx1250)
__device__ __forceinline__ unsigned pk2bf(float a, float b) {
  v2f t; t.x = a; t.y = b;
  v2bf r = __builtin_convertvector(t, v2bf);
  union { v2bf v; unsigned u; } c; c.v = r;
  return c.u;
}
__device__ __forceinline__ float wred32(float v) {
  #pragma unroll
  for (int m = 16; m > 0; m >>= 1) v += __shfl_xor(v, m, 32);
  return v;
}

// ---- CDNA5 async global->LDS copy (ASYNCcnt path) -------------------------
__device__ __forceinline__ void async_cp16(const void* g, void* l) {
#if __has_builtin(__builtin_amdgcn_global_load_async_to_lds_b128)
  __builtin_amdgcn_global_load_async_to_lds_b128(
      (__attribute__((address_space(1))) v4i*)g,
      (__attribute__((address_space(3))) v4i*)l, 0, 0);
#else
  unsigned lo = (unsigned)(unsigned long long)
      (__attribute__((address_space(3))) void*)l;
  asm volatile("global_load_async_to_lds_b128 %0, %1, off"
               :: "v"(lo), "v"((unsigned long long)g) : "memory");
#endif
}
__device__ __forceinline__ void wait_async0() {
#if __has_builtin(__builtin_amdgcn_s_wait_asynccnt)
  __builtin_amdgcn_s_wait_asynccnt(0);
#else
  asm volatile("s_wait_asynccnt 0" ::: "memory");
#endif
}
// issue this block's share of a global->LDS weight stage (16B granules)
__device__ __forceinline__ void stage_async(const unsigned short* __restrict__ g,
                                            unsigned short* l, int nbytes, int tid) {
  const char* gc = (const char*)g;
  char* lc = (char*)l;
  for (int off = tid * 16; off < nbytes; off += THREADS * 16)
    async_cp16(gc + off, lc + off);
}

// ---------------------------------------------------------------------------
// Prep: swizzle f32 weight W (K x N, row-major) into WMMA bf16 B-fragments.
// Fragment (nt,kt): per lane L (0..31), 16 bf16 slots s.
//   n = nt*16 + (L&15);  g = L>>4;  kk = s + g*8 + (s>=8 ? 8 : 0)
// Stored contiguously: dst[((nt*KT+kt)*32 + L)*16 + s]  -> 32B/lane loads.
// ---------------------------------------------------------------------------
extern "C" __global__ void swizzle_w(const float* __restrict__ W,
                                     unsigned short* __restrict__ dst,
                                     int K, int N) {
  int KT = K >> 5, NT = N >> 4;
  int total = NT * KT * 512;
  for (int idx = blockIdx.x * blockDim.x + threadIdx.x; idx < total;
       idx += gridDim.x * blockDim.x) {
    int s    = idx & 15;
    int L    = (idx >> 4) & 31;
    int frag = idx >> 9;
    int kt = frag % KT, nt = frag / KT;
    int g  = L >> 4;
    int n  = nt * 16 + (L & 15);
    int kk = s + g * 8 + ((s >= 8) ? 8 : 0);
    dst[idx] = f2bf(W[(kt * 32 + kk) * N + n]);
  }
}

// ---------------------------------------------------------------------------
// 16 x (KT*32)  @  (KT*32) x 256 GEMM for one wave.
// A from per-wave LDS (row-major bf16, stride CH), B fragments from the
// block-shared LDS weight stage, result written back to LDS row-major bf16.
// ---------------------------------------------------------------------------
union AF { unsigned u[8]; v16bf v; };

template<int KT>
__device__ __forceinline__ void gemm_tile(const unsigned short* A,
                                          const unsigned short* Bw,
                                          unsigned short* Out, int lane) {
  const int M = lane & 15, gq = lane >> 4;
  AF af[KT];
  #pragma unroll
  for (int kt = 0; kt < KT; ++kt) {
    #pragma unroll
    for (int r = 0; r < 8; ++r) {
      int kk = 2 * r + gq * 8 + ((r >= 4) ? 8 : 0);   // consecutive K pair
      af[kt].u[r] = *(const unsigned*)(A + M * CH + kt * 32 + kk);
    }
  }
  for (int nt = 0; nt < 16; ++nt) {
    v8f acc = {};
    #pragma unroll
    for (int kt = 0; kt < KT; ++kt) {
      v16bf b = *(const v16bf*)(Bw + ((nt * KT + kt) * 32 + lane) * 16);
      acc = __builtin_amdgcn_wmma_f32_16x16x32_bf16(
          false, af[kt].v, false, b, (short)0, acc, false, false);
    }
    // packed f32->bf16 spill; rows r and r+1 get lo/hi halves
    #pragma unroll
    for (int r = 0; r < 8; r += 2) {
      unsigned pk = pk2bf(acc[r], acc[r + 1]);
      Out[(r     + gq * 8) * CH + nt * 16 + M] = (unsigned short)pk;
      Out[(r + 1 + gq * 8) * CH + nt * 16 + M] = (unsigned short)(pk >> 16);
    }
  }
}

// incoming neighbors on the 4x4 grid (+self loop)
__device__ __forceinline__ int nbr(int dst, int slot, bool& valid) {
  int i = dst >> 2, j = dst & 3;
  switch (slot) {
    case 0: valid = true;   return dst;
    case 1: valid = (j > 0); return valid ? dst - 1 : dst;
    case 2: valid = (j < 3); return valid ? dst + 1 : dst;
    case 3: valid = (i > 0); return valid ? dst - 4 : dst;
    default: valid = (i < 3); return valid ? dst + 4 : dst;
  }
}

// ---------------------------------------------------------------------------
// GAT attention: softmax over incoming edges per (dst, head), aggregate.
// FINAL=false: out[d][h*64+f] = relu(sum + bias), written bf16 to `out`.
// FINAL=true : head-mean + node-mean fused -> gvec[64] (+bias64), no relu.
// ---------------------------------------------------------------------------
template<bool FINAL>
__device__ void attn(const unsigned short* xp, unsigned short* out,
                     const float* __restrict__ a_src, const float* __restrict__ a_dst,
                     const float* __restrict__ bias,
                     float* es, float* ed, float* al, float* gvec, int lane) {
  // per-node per-head src/dst attention scalars (paired u32 LDS reads)
  #pragma unroll
  for (int p = 0; p < 2; ++p) {
    int idx = lane + 32 * p;
    int n = idx & 15, h = idx >> 4;
    float s0 = 0.f, s1 = 0.f;
    for (int f = 0; f < 64; f += 2) {
      unsigned u = *(const unsigned*)(xp + n * CH + h * 64 + f);
      float x0 = __uint_as_float(u << 16);
      float x1 = __uint_as_float(u & 0xffff0000u);
      s0 += x0 * a_src[h * 64 + f] + x1 * a_src[h * 64 + f + 1];
      s1 += x0 * a_dst[h * 64 + f] + x1 * a_dst[h * 64 + f + 1];
    }
    es[n * 4 + h] = s0;
    ed[n * 4 + h] = s1;
  }
  __syncthreads();
  // softmax over incoming edges
  #pragma unroll
  for (int p = 0; p < 2; ++p) {
    int idx = lane + 32 * p;
    int d = idx & 15, h = idx >> 4;
    float edv = ed[d * 4 + h];
    float l[5]; bool v[5]; float mx = -1e30f;
    #pragma unroll
    for (int s = 0; s < 5; ++s) {
      bool ok; int src = nbr(d, s, ok);
      float lg = es[src * 4 + h] + edv;
      lg = lg > 0.f ? lg : 0.2f * lg;        // leaky_relu(0.2)
      v[s] = ok; l[s] = lg;
      if (ok && lg > mx) mx = lg;
    }
    float den = 0.f; float w[5];
    #pragma unroll
    for (int s = 0; s < 5; ++s) { w[s] = v[s] ? __expf(l[s] - mx) : 0.f; den += w[s]; }
    float inv = 1.f / den;
    #pragma unroll
    for (int s = 0; s < 5; ++s) al[d * 20 + h * 5 + s] = w[s] * inv;
  }
  __syncthreads();
  // weighted aggregation; each lane owns feature columns {lane, lane+32}
  float g0 = 0.f, g1 = 0.f;
  for (int d = 0; d < 16; ++d) {
    #pragma unroll
    for (int h = 0; h < 4; ++h) {
      float a0 = 0.f, a1 = 0.f;
      #pragma unroll
      for (int s = 0; s < 5; ++s) {
        bool ok; int src = nbr(d, s, ok);
        float a = al[d * 20 + h * 5 + s];
        a0 += a * bf2f(xp[src * CH + h * 64 + lane]);
        a1 += a * bf2f(xp[src * CH + h * 64 + 32 + lane]);
      }
      if (FINAL) {
        g0 += a0; g1 += a1;                  // fuse head-mean + node-mean
      } else {
        int c0 = h * 64 + lane, c1 = c0 + 32;
        float r0 = a0 + bias[c0]; r0 = r0 > 0.f ? r0 : 0.f;
        float r1 = a1 + bias[c1]; r1 = r1 > 0.f ? r1 : 0.f;
        unsigned pk = pk2bf(r0, r1);
        out[d * CH + c0] = (unsigned short)pk;
        out[d * CH + c1] = (unsigned short)(pk >> 16);
      }
    }
  }
  if (FINAL) {
    gvec[lane]      = g0 * (1.f / 64.f) + bias[lane];
    gvec[lane + 32] = g1 * (1.f / 64.f) + bias[lane + 32];
  }
  __syncthreads();
}

// ---------------------------------------------------------------------------
// Main kernel: 8 waves / block, one batch element per wave; weights staged
// per block in LDS via async DMA, overlapped with attention compute.
// ---------------------------------------------------------------------------
extern "C" __global__ void __launch_bounds__(THREADS, 1)
gat_main(const float* __restrict__ x,
         const float* __restrict__ in_w, const float* __restrict__ in_b,
         const unsigned short* __restrict__ w0f, const float* __restrict__ a0s,
         const float* __restrict__ a0d, const float* __restrict__ b0,
         const unsigned short* __restrict__ w1f, const float* __restrict__ a1s,
         const float* __restrict__ a1d, const float* __restrict__ b1,
         const unsigned short* __restrict__ w2f, const float* __restrict__ a2s,
         const float* __restrict__ a2d, const float* __restrict__ b2,
         const float* __restrict__ mw1, const float* __restrict__ mb1,
         const float* __restrict__ lg1, const float* __restrict__ lb1,
         const float* __restrict__ mw2, const float* __restrict__ mb2,
         const float* __restrict__ lg2, const float* __restrict__ lb2,
         float* __restrict__ outp) {
  extern __shared__ char smem[];
  const int tid  = threadIdx.x;
  const int lane = tid & 31;
  const int wid  = tid >> 5;
  const int b    = blockIdx.x * WAVES_PER_BLOCK + wid;

  unsigned short* wstage = (unsigned short*)smem;          // 128KB shared stage
  char* base = smem + WSTAGE_BYTES + wid * LDS_PER_WAVE_BYTES;
  unsigned short* bufA = (unsigned short*)base;            // 16x256 bf16
  unsigned short* bufB = bufA + 16 * CH;                   // 16x256 bf16
  float* nd = (float*)(bufB + 16 * CH);                    // 256 (x-layout [c][n])
  float* es = nd + 256;                                    // 64
  float* ed = es + 64;                                     // 64
  float* al = ed + 64;                                     // 320
  float* gv = al + 320;                                    // 64
  float* yv = gv + 64;                                     // 128

  // ---- stage nodes (B,C,4,4): flat b*256 + c*16 + n ----
  const float* xb = x + (size_t)b * 256;
  #pragma unroll
  for (int t = 0; t < 2; ++t)
    *(float4*)(nd + lane * 8 + t * 4) = *(const float4*)(xb + lane * 8 + t * 4);
  __syncthreads();

  // kick off async stage of GAT0 weights; overlaps with h0 compute
  stage_async(w0f, wstage, 32768, tid);

  // ---- h0 = relu(nodes @ in_w + in_b): 16x16 @ 16x64 ----
  for (int n = 0; n < 16; ++n) {
    int c0 = lane, c1 = lane + 32;
    float acc0 = in_b[c0], acc1 = in_b[c1];
    #pragma unroll
    for (int k = 0; k < 16; ++k) {
      float xv = nd[k * 16 + n];
      acc0 += xv * in_w[k * 64 + c0];
      acc1 += xv * in_w[k * 64 + c1];
    }
    acc0 = acc0 > 0.f ? acc0 : 0.f;
    acc1 = acc1 > 0.f ? acc1 : 0.f;
    unsigned pk = pk2bf(acc0, acc1);
    bufA[n * CH + c0] = (unsigned short)pk;
    bufA[n * CH + c1] = (unsigned short)(pk >> 16);
  }
  wait_async0();
  __syncthreads();

  // ---- GAT0 (K=64) ----
  gemm_tile<2>(bufA, wstage, bufB, lane);
  __syncthreads();                          // all waves done with wstage
  stage_async(w1f, wstage, WSTAGE_BYTES, tid);   // DMA GAT1 weights under attn
  attn<false>(bufB, bufA, a0s, a0d, b0, es, ed, al, gv, lane);
  wait_async0();
  __syncthreads();

  // ---- GAT1 (K=256) ----
  gemm_tile<8>(bufA, wstage, bufB, lane);
  __syncthreads();
  stage_async(w2f, wstage, WSTAGE_BYTES, tid);   // DMA GAT2 weights under attn
  attn<false>(bufB, bufA, a1s, a1d, b1, es, ed, al, gv, lane);
  wait_async0();
  __syncthreads();

  // ---- GAT2 (K=256, concat=False -> gvec) ----
  gemm_tile<8>(bufA, wstage, bufB, lane);
  __syncthreads();
  attn<true>(bufB, bufA, a2s, a2d, b2, es, ed, al, gv, lane);

  // ---- MLP1: y = relu(LN(g @ mw1 + mb1)) , 64 -> 128 ----
  float y[4];
  #pragma unroll
  for (int t = 0; t < 4; ++t) {
    int k = lane + 32 * t;
    float acc = mb1[k];
    for (int f = 0; f < 64; ++f) acc += gv[f] * mw1[f * 128 + k];
    y[t] = acc;
  }
  float mean = wred32(y[0] + y[1] + y[2] + y[3]) * (1.f / 128.f);
  float vs = 0.f;
  #pragma unroll
  for (int t = 0; t < 4; ++t) { float d = y[t] - mean; vs += d * d; }
  float rstd = __frsqrt_rn(wred32(vs) * (1.f / 128.f) + 1e-5f);
  #pragma unroll
  for (int t = 0; t < 4; ++t) {
    int k = lane + 32 * t;
    float v2 = (y[t] - mean) * rstd * lg1[k] + lb1[k];
    yv[k] = v2 > 0.f ? v2 : 0.f;
  }
  __syncthreads();

  // ---- MLP2: out = relu(LN(y @ mw2 + mb2)) , 128 -> 256 ----
  float o[8];
  #pragma unroll
  for (int t = 0; t < 8; ++t) {
    int c = lane + 32 * t;
    float acc = mb2[c];
    for (int k = 0; k < 128; ++k) acc += yv[k] * mw2[k * 256 + c];
    o[t] = acc;
  }
  float s2 = 0.f;
  #pragma unroll
  for (int t = 0; t < 8; ++t) s2 += o[t];
  float mean2 = wred32(s2) * (1.f / 256.f);
  float vs2 = 0.f;
  #pragma unroll
  for (int t = 0; t < 8; ++t) { float d = o[t] - mean2; vs2 += d * d; }
  float rstd2 = __frsqrt_rn(wred32(vs2) * (1.f / 256.f) + 1e-5f);
  #pragma unroll
  for (int t = 0; t < 8; ++t) {
    int c = lane + 32 * t;
    float v2 = (o[t] - mean2) * rstd2 * lg2[c] + lb2[c];
    outp[(size_t)b * 256 + c] = v2 > 0.f ? v2 : 0.f;
  }
}

// ---------------------------------------------------------------------------
extern "C" void kernel_launch(void* const* d_in, const int* in_sizes, int n_in,
                              void* d_out, int out_size, void* d_ws, size_t ws_size,
                              hipStream_t stream) {
  (void)n_in; (void)out_size; (void)ws_size;
  unsigned short* ws  = (unsigned short*)d_ws;
  unsigned short* w0f = ws;                      // 2*16*512   = 16384 bf16
  unsigned short* w1f = w0f + 16384;             // 8*16*512   = 65536 bf16
  unsigned short* w2f = w1f + 65536;             // 8*16*512   = 65536 bf16

  swizzle_w<<<64, 256, 0, stream>>>((const float*)d_in[3],  w0f, 64, 256);
  swizzle_w<<<256, 256, 0, stream>>>((const float*)d_in[7],  w1f, 256, 256);
  swizzle_w<<<256, 256, 0, stream>>>((const float*)d_in[11], w2f, 256, 256);

  // unconditional + deterministic (not a stream op; capture-safe)
  (void)hipFuncSetAttribute((const void*)gat_main,
                            hipFuncAttributeMaxDynamicSharedMemorySize,
                            SMEM_TOTAL);

  int B = in_sizes[0] / 256;                     // 4096
  gat_main<<<B / WAVES_PER_BLOCK, THREADS, SMEM_TOTAL, stream>>>(
      (const float*)d_in[0],
      (const float*)d_in[1],  (const float*)d_in[2],
      w0f, (const float*)d_in[4],  (const float*)d_in[5],  (const float*)d_in[6],
      w1f, (const float*)d_in[8],  (const float*)d_in[9],  (const float*)d_in[10],
      w2f, (const float*)d_in[12], (const float*)d_in[13], (const float*)d_in[14],
      (const float*)d_in[15], (const float*)d_in[16],
      (const float*)d_in[17], (const float*)d_in[18],
      (const float*)d_in[19], (const float*)d_in[20],
      (const float*)d_in[21], (const float*)d_in[22],
      (float*)d_out);
}
